// KoLeoLoss_11330123726868
// MI455X (gfx1250) — matrix operand order repaired and checked by reading, not compiled, not run on hardware
//
#include <hip/hip_runtime.h>
#include <hip/hip_bf16.h>

typedef __attribute__((ext_vector_type(16))) _Float16 v16h;
typedef __attribute__((ext_vector_type(8)))  _Float16 v8h;
typedef __attribute__((ext_vector_type(4)))  _Float16 v4h;
typedef __attribute__((ext_vector_type(8)))  float    v8f;

#define NB       4096      // batch rows per view
#define NVIEWS   2
#define NDIM     1024
#define KEPS     1e-8f
#define KTILE    32
#define LDS_PITCH 40       // halves per LDS row (32 data + 8 pad -> conflict-free)

// ---------------------------------------------------------------------------
// Kernel 1: L2-normalize each (b,v) row; emit f16 copy in [view][row][dim]
// layout for the GEMM, plus n2 = ||x_hat||^2 = s/(s+eps) per row.
// ---------------------------------------------------------------------------
__global__ __launch_bounds__(256) void koleo_normalize(
    const float* __restrict__ x, _Float16* __restrict__ xh,
    float* __restrict__ n2)
{
    const int wave = blockIdx.x * 8 + (threadIdx.x >> 5);   // 0..8191
    const int lane = threadIdx.x & 31;
    const int v = wave >> 12;          // view
    const int b = wave & (NB - 1);     // row in view

    const float4* src = (const float4*)(x + ((size_t)b * NVIEWS + v) * NDIM);
    float4 vals[8];
    float s = 0.f;
#pragma unroll
    for (int i = 0; i < 8; ++i) {
        float4 f = src[i * 32 + lane];
        vals[i] = f;
        s += f.x * f.x + f.y * f.y + f.z * f.z + f.w * f.w;
    }
#pragma unroll
    for (int off = 16; off > 0; off >>= 1) s += __shfl_xor(s, off, 32);

    const float inv = 1.0f / sqrtf(s + KEPS);
    v4h* dst = (v4h*)(xh + ((size_t)v * NB + b) * NDIM);
#pragma unroll
    for (int i = 0; i < 8; ++i) {
        float4 f = vals[i];
        v4h h;
        h.x = (_Float16)(f.x * inv); h.y = (_Float16)(f.y * inv);
        h.z = (_Float16)(f.z * inv); h.w = (_Float16)(f.w * inv);
        dst[i * 32 + lane] = h;
    }
    if (lane == 0) n2[wave] = s / (s + KEPS);   // index == v*NB + b
}

// ---------------------------------------------------------------------------
// Kernel 2: tiled X * X^T via v_wmma_f32_16x16x32_f16, double-buffered LDS,
// branchless software pipeline, per-row argmax -> packed u64 atomicMax.
// ---------------------------------------------------------------------------
__device__ __forceinline__ unsigned long long shfl_xor_u64(unsigned long long x, int m) {
    unsigned int lo = (unsigned int)x, hi = (unsigned int)(x >> 32);
    lo = __shfl_xor(lo, m, 32);
    hi = __shfl_xor(hi, m, 32);
    return ((unsigned long long)hi << 32) | lo;
}

__device__ __forceinline__ v16h frag_cat(v8h lo, v8h hi) {
    return __builtin_shufflevector(lo, hi,
        0, 1, 2, 3, 4, 5, 6, 7, 8, 9, 10, 11, 12, 13, 14, 15);
}

#define WMMA_F16(a, b, c) __builtin_amdgcn_wmma_f32_16x16x32_f16( \
    false, (a), false, (b), (short)0, (c), false, false)

// One pipelined K-step: stage the global tile at kNext into (lAs,lBs) while
// computing from (lAc,lBc). All staging loads issue first so the entire
// fragment-load + 8-WMMA block covers their latency; stores + barrier last.
__device__ __forceinline__ void gemm_phase(
    const _Float16* __restrict__ gA, const _Float16* __restrict__ gB, int kNext,
    const _Float16* lAc, const _Float16* lBc,
    _Float16* lAs, _Float16* lBs,
    int srow, int sseg, int arow0, int akb, int brow, int bkb, int waveN,
    v8f (&acc)[2][4])
{
    uint4 ra0 = *(const uint4*)(gA + kNext);
    uint4 ra1 = *(const uint4*)(gA + kNext + 8);
    uint4 rb0 = *(const uint4*)(gB + kNext);
    uint4 rb1 = *(const uint4*)(gB + kNext + 8);
    if (kNext + 2 * KTILE < NDIM) {
        __builtin_prefetch(gA + kNext + 2 * KTILE, 0, 1);
        __builtin_prefetch(gB + kNext + 2 * KTILE, 0, 1);
    }

    v16h af[2];
#pragma unroll
    for (int mi = 0; mi < 2; ++mi) {
        const _Float16* p = lAc + (arow0 + mi * 16) * LDS_PITCH + akb;
        af[mi] = frag_cat(*(const v8h*)p, *(const v8h*)(p + 16));
    }
    v16h bf[4];
#pragma unroll
    for (int nbt = 0; nbt < 4; ++nbt) {
        const _Float16* p = lBc + (waveN * 64 + nbt * 16 + brow) * LDS_PITCH + bkb;
        bf[nbt] = frag_cat(*(const v8h*)p, *(const v8h*)(p + 16));
    }
#pragma unroll
    for (int nbt = 0; nbt < 4; ++nbt) {
        acc[0][nbt] = WMMA_F16(af[0], bf[nbt], acc[0][nbt]);
        acc[1][nbt] = WMMA_F16(af[1], bf[nbt], acc[1][nbt]);
    }

    *(uint4*)&lAs[srow * LDS_PITCH + sseg]     = ra0;
    *(uint4*)&lAs[srow * LDS_PITCH + sseg + 8] = ra1;
    *(uint4*)&lBs[srow * LDS_PITCH + sseg]     = rb0;
    *(uint4*)&lBs[srow * LDS_PITCH + sseg + 8] = rb1;
    __syncthreads();
}

// compute-only K-step (no staging) for the pipeline epilogue
__device__ __forceinline__ void gemm_phase_tail(
    const _Float16* lAc, const _Float16* lBc,
    int arow0, int akb, int brow, int bkb, int waveN, v8f (&acc)[2][4])
{
    v16h af[2];
#pragma unroll
    for (int mi = 0; mi < 2; ++mi) {
        const _Float16* p = lAc + (arow0 + mi * 16) * LDS_PITCH + akb;
        af[mi] = frag_cat(*(const v8h*)p, *(const v8h*)(p + 16));
    }
#pragma unroll
    for (int nbt = 0; nbt < 4; ++nbt) {
        const _Float16* p = lBc + (waveN * 64 + nbt * 16 + brow) * LDS_PITCH + bkb;
        v16h bf = frag_cat(*(const v8h*)p, *(const v8h*)(p + 16));
        acc[0][nbt] = WMMA_F16(af[0], bf, acc[0][nbt]);
        acc[1][nbt] = WMMA_F16(af[1], bf, acc[1][nbt]);
    }
}

__global__ __launch_bounds__(256)
__attribute__((amdgpu_waves_per_eu(2, 2)))   // 8 waves/WG == 2/SIMD exactly:
                                             // stop occupancy-driven pressure
                                             // sinking, unlock VGPR budget
void koleo_argmax_gemm(
    const _Float16* __restrict__ xh, unsigned long long* __restrict__ best)
{
    __shared__ _Float16 ldsA[2][128 * LDS_PITCH];   // 2 x 10 KB
    __shared__ _Float16 ldsB[2][128 * LDS_PITCH];   // 2 x 10 KB

    const int v  = blockIdx.z;
    const int m0 = blockIdx.y * 128;
    const int n0 = blockIdx.x * 128;
    const _Float16* X = xh + (size_t)v * NB * NDIM;

    const int tid   = threadIdx.x;
    const int lane  = tid & 31;
    const int wv    = tid >> 5;
    const int waveM = wv & 3;        // 32-row strip
    const int waveN = wv >> 2;       // 64-col half

    // cooperative staging: thread t handles 16 halves of row (t>>1)
    const int srow = tid >> 1;
    const int sseg = (tid & 1) * 16;
    const _Float16* gA = X + (size_t)(m0 + srow) * NDIM + sseg;
    const _Float16* gB = X + (size_t)(n0 + srow) * NDIM + sseg;

    // fragment layouts per ISA 7.12.2 (16-bit A 16x32, 16-bit B 32x16)
    const int arow0 = waveM * 32 + (lane & 15);
    const int akb   = (lane >> 4) * 8;   // A: lanes<16 -> K 0..7/16..23, else 8..15/24..31
    const int bkb   = (lane >> 4) * 8;   // B: lanes<16 -> K 0..7/16..23, else 8..15/24..31
    const int brow  = lane & 15;         // B: N = lane&15

    v8f acc[2][4] = {};

    // prologue: tile 0 -> buffer 0
    {
        uint4 r0 = *(const uint4*)(gA);
        uint4 r1 = *(const uint4*)(gA + 8);
        uint4 r2 = *(const uint4*)(gB);
        uint4 r3 = *(const uint4*)(gB + 8);
        *(uint4*)&ldsA[0][srow * LDS_PITCH + sseg]     = r0;
        *(uint4*)&ldsA[0][srow * LDS_PITCH + sseg + 8] = r1;
        *(uint4*)&ldsB[0][srow * LDS_PITCH + sseg]     = r2;
        *(uint4*)&ldsB[0][srow * LDS_PITCH + sseg + 8] = r3;
    }
    __syncthreads();

    // steady state: tiles 0..29 computed, tiles 1..30 staged (all unconditional)
#pragma unroll 1
    for (int kt2 = 0; kt2 < 15; ++kt2) {
        gemm_phase(gA, gB, (2 * kt2 + 1) * KTILE,
                   ldsA[0], ldsB[0], ldsA[1], ldsB[1],
                   srow, sseg, arow0, akb, brow, bkb, waveN, acc);
        gemm_phase(gA, gB, (2 * kt2 + 2) * KTILE,
                   ldsA[1], ldsB[1], ldsA[0], ldsB[0],
                   srow, sseg, arow0, akb, brow, bkb, waveN, acc);
    }
    // epilogue: compute tile 30 while staging tile 31, then compute tile 31
    gemm_phase(gA, gB, 31 * KTILE,
               ldsA[0], ldsB[0], ldsA[1], ldsB[1],
               srow, sseg, arow0, akb, brow, bkb, waveN, acc);
    gemm_phase_tail(ldsA[1], ldsB[1], arow0, akb, brow, bkb, waveN, acc);

    // C/D layout: VGPR r -> M = r + 8*(lane>>4), N = lane&15
    const int mbase = m0 + waveM * 32 + (lane >> 4) * 8;
    const int nbase = n0 + waveN * 64 + (lane & 15);
#pragma unroll
    for (int mi = 0; mi < 2; ++mi) {
#pragma unroll
        for (int r = 0; r < 8; ++r) {
            const int m = mbase + mi * 16 + r;
            unsigned long long key = 0ull;
#pragma unroll
            for (int nbt = 0; nbt < 4; ++nbt) {
                const int n = nbase + nbt * 16;
                if (n == m) continue;                       // diagonal excluded
                unsigned int fb  = __float_as_uint(acc[mi][nbt][r]);
                unsigned int ord = (fb & 0x80000000u) ? ~fb : (fb | 0x80000000u);
                unsigned long long k = ((unsigned long long)ord << 32) | (unsigned int)n;
                key = (k > key) ? k : key;
            }
            // reduce across the 16 lanes sharing this m
#pragma unroll
            for (int off = 8; off > 0; off >>= 1) {
                unsigned long long o = shfl_xor_u64(key, off);
                key = (o > key) ? o : key;
            }
            if ((lane & 15) == 0)
                atomicMax(best + (size_t)v * NB + m, key);
        }
    }
}

// ---------------------------------------------------------------------------
// Kernel 3: dist = sqrt(n2[b] + n2[nn] - 2*dot); loss = sum_v mean_b -log(d+eps)
// ---------------------------------------------------------------------------
__global__ __launch_bounds__(256) void koleo_finalize(
    const unsigned long long* __restrict__ best,
    const float* __restrict__ n2, float* __restrict__ out)
{
    __shared__ float red[256];
    float s = 0.f;
    for (int i = threadIdx.x; i < NVIEWS * NB; i += 256) {
        unsigned long long key = best[i];
        unsigned int ord = (unsigned int)(key >> 32);
        unsigned int n   = (unsigned int)(key & 0xffffffffu);
        unsigned int fb  = (ord & 0x80000000u) ? (ord & 0x7fffffffu) : ~ord;
        float dot = __uint_as_float(fb);
        float d2  = n2[i] + n2[(i & ~(NB - 1)) + (int)n] - 2.0f * dot;
        d2 = d2 > 0.f ? d2 : 0.f;
        s += -logf(sqrtf(d2) + KEPS);
    }
    red[threadIdx.x] = s;
    __syncthreads();
    for (int off = 128; off > 0; off >>= 1) {
        if (threadIdx.x < off) red[threadIdx.x] += red[threadIdx.x + off];
        __syncthreads();
    }
    if (threadIdx.x == 0) out[0] = red[0] * (1.0f / (float)NB);
}

__global__ void koleo_init(unsigned long long* best) {
    int i = blockIdx.x * blockDim.x + threadIdx.x;
    if (i < NVIEWS * NB) best[i] = 0ull;
}

// ---------------------------------------------------------------------------
extern "C" void kernel_launch(void* const* d_in, const int* in_sizes, int n_in,
                              void* d_out, int out_size, void* d_ws, size_t ws_size,
                              hipStream_t stream)
{
    const float* x = (const float*)d_in[0];
    float* out = (float*)d_out;

    char* ws = (char*)d_ws;
    _Float16* xh = (_Float16*)ws;                                   // 16 MB
    size_t off_xh = (size_t)NVIEWS * NB * NDIM * sizeof(_Float16);
    float* n2 = (float*)(ws + off_xh);                              // 32 KB
    size_t off_n2 = off_xh + (size_t)NVIEWS * NB * sizeof(float);
    unsigned long long* best = (unsigned long long*)(ws + off_n2);  // 64 KB

    koleo_init<<<dim3((NVIEWS * NB + 255) / 256), dim3(256), 0, stream>>>(best);
    koleo_normalize<<<dim3(NVIEWS * NB / 8), dim3(256), 0, stream>>>(x, xh, n2);
    koleo_argmax_gemm<<<dim3(NB / 128, NB / 128, NVIEWS), dim3(256), 0, stream>>>(xh, best);
    koleo_finalize<<<dim3(1), dim3(256), 0, stream>>>(best, n2, out);
}
